// FeaturePropagate_2173253452311
// MI455X (gfx1250) — matrix operand order repaired and checked by reading, not compiled, not run on hardware
//
#include <hip/hip_runtime.h>

typedef __attribute__((ext_vector_type(16))) __bf16 v16bf;
typedef __attribute__((ext_vector_type(8)))  float  v8f;

#define NB 2
#define NQ 16384
#define NS 4096
#define CQF 128
#define CSF 256
#define CIN 384
#define O1 256
#define O2 256

// ---------------- f32 -> bf16 convert (weights) ----------------
__global__ void k_f32_to_bf16(const float* __restrict__ in, __bf16* __restrict__ out, int n) {
    int i = blockIdx.x * blockDim.x + threadIdx.x;
    if (i < n) out[i] = (__bf16)in[i];
}

// ---------------- 3-NN: LDS-resident source points, top-3 scan ----------------
__global__ void k_three_nn(const float* __restrict__ qp, const float* __restrict__ sp,
                           int* __restrict__ nn_idx, float* __restrict__ nn_w) {
    __shared__ float sx[NS], sy[NS], sz[NS];
    const int b = blockIdx.y;
    const int q = blockIdx.x * blockDim.x + threadIdx.x;
    const float* spb = sp + (size_t)b * 3 * NS;
    for (int i = threadIdx.x; i < NS; i += blockDim.x) {
        sx[i] = spb[i];
        sy[i] = spb[NS + i];
        sz[i] = spb[2 * NS + i];
    }
    __syncthreads();
    const float* qpb = qp + (size_t)b * 3 * NQ;
    const float qx = qpb[q], qy = qpb[NQ + q], qz = qpb[2 * NQ + q];
    float d0 = 3.0e38f, d1 = 3.0e38f, d2 = 3.0e38f;
    int   i0 = 0, i1 = 0, i2 = 0;
    for (int s = 0; s < NS; ++s) {
        float dx = qx - sx[s], dy = qy - sy[s], dz = qz - sz[s];
        float d = dx * dx + dy * dy + dz * dz;
        if (d < d2) {
            if (d < d1) {
                d2 = d1; i2 = i1;
                if (d < d0) { d1 = d0; i1 = i0; d0 = d; i0 = s; }
                else        { d1 = d;  i1 = s; }
            } else { d2 = d; i2 = s; }
        }
    }
    float w0 = 1.f / (fmaxf(d0, 0.f) + 1e-5f);
    float w1 = 1.f / (fmaxf(d1, 0.f) + 1e-5f);
    float w2 = 1.f / (fmaxf(d2, 0.f) + 1e-5f);
    float inv = 1.f / (w0 + w1 + w2);
    size_t base = ((size_t)b * NQ + q) * 3;
    nn_idx[base] = i0; nn_idx[base + 1] = i1; nn_idx[base + 2] = i2;
    nn_w[base] = w0 * inv; nn_w[base + 1] = w1 * inv; nn_w[base + 2] = w2 * inv;
}

// ------ interpolate + concat -> bf16 transposed activations (B, Q, CIN) ------
__global__ void k_interp_concat(const float* __restrict__ qf, const float* __restrict__ sf,
                                const int* __restrict__ nn_idx, const float* __restrict__ nn_w,
                                __bf16* __restrict__ Xt) {
    const int q = blockIdx.x;
    const int b = blockIdx.y;
    const int t = threadIdx.x;  // 0..383
    __shared__ int   si[3];
    __shared__ float sw[3];
    if (t < 3) {
        size_t base = ((size_t)b * NQ + q) * 3;
        si[t] = nn_idx[base + t];
        sw[t] = nn_w[base + t];
    }
    __syncthreads();
    __bf16* out = Xt + ((size_t)b * NQ + q) * CIN;
    if (t < CQF) {
        out[t] = (__bf16)qf[((size_t)b * CQF + t) * NQ + q];
    } else {
        int c = t - CQF;
        const float* sfr = sf + ((size_t)b * CSF + c) * NS;
        float v = sw[0] * sfr[si[0]] + sw[1] * sfr[si[1]] + sw[2] * sfr[si[2]];
        out[t] = (__bf16)v;
    }
}

// ---------------- bf16 WMMA GEMM: Y[b,m,n] = sum_k W[m,k] * Xt[b,n,k] + bias[m] ----------------
// One wave (32 threads) per block computes a 16(M) x 64(N) tile, K-loop in steps of 32.
__global__ void k_gemm_bf16(const __bf16* __restrict__ Wm, const __bf16* __restrict__ Xt,
                            const float* __restrict__ bias, float* __restrict__ Y,
                            int O, int C, int Qn) {
    const int lane = threadIdx.x & 31;
    const int half = lane >> 4;
    const int mn   = lane & 15;
    const int n0 = blockIdx.x * 64;
    const int m0 = blockIdx.y * 16;
    const int b  = blockIdx.z;

    v8f acc0 = {}, acc1 = {}, acc2 = {}, acc3 = {};

    const unsigned* wrow  = (const unsigned*)(Wm + (size_t)(m0 + mn) * C);
    const size_t    xbase = (size_t)b * Qn * C;
    const unsigned* xr0 = (const unsigned*)(Xt + xbase + (size_t)(n0 +  0 + mn) * C);
    const unsigned* xr1 = (const unsigned*)(Xt + xbase + (size_t)(n0 + 16 + mn) * C);
    const unsigned* xr2 = (const unsigned*)(Xt + xbase + (size_t)(n0 + 32 + mn) * C);
    const unsigned* xr3 = (const unsigned*)(Xt + xbase + (size_t)(n0 + 48 + mn) * C);

    for (int k0 = 0; k0 < C; k0 += 32) {
        const int kd = k0 >> 1;  // dword offset within row
        union { v16bf v; unsigned u[8]; } a, b0, b1, b2, b3;
        // A as loaded here (contiguous k order, same as B): VGPR v, lane-half h: K = h*16 + v*2 (+0/+1)
        #pragma unroll
        for (int v = 0; v < 8; ++v)
            a.u[v] = wrow[kd + half * 8 + v];
        // B as loaded here (32x16, contiguous k order, same as A): lanes 0-15 K=0..15, lanes 16-31 K=16..31; VGPR v = K pair (2v,2v+1)
        #pragma unroll
        for (int v = 0; v < 8; ++v) {
            int off = kd + half * 8 + v;
            b0.u[v] = xr0[off];
            b1.u[v] = xr1[off];
            b2.u[v] = xr2[off];
            b3.u[v] = xr3[off];
        }
        acc0 = __builtin_amdgcn_wmma_f32_16x16x32_bf16(false, a.v, false, b0.v, (short)0, acc0, false, false);
        acc1 = __builtin_amdgcn_wmma_f32_16x16x32_bf16(false, a.v, false, b1.v, (short)0, acc1, false, false);
        acc2 = __builtin_amdgcn_wmma_f32_16x16x32_bf16(false, a.v, false, b2.v, (short)0, acc2, false, false);
        acc3 = __builtin_amdgcn_wmma_f32_16x16x32_bf16(false, a.v, false, b3.v, (short)0, acc3, false, false);
    }
    // C/D layout: VGPR r -> M = m0 + half*8 + r, N = n0 + subtile*16 + mn
    #pragma unroll
    for (int r = 0; r < 8; ++r) {
        int m = m0 + half * 8 + r;
        float bv = bias[m];
        float* yr = Y + ((size_t)b * O + m) * Qn + n0 + mn;
        yr[0]  = acc0[r] + bv;
        yr[16] = acc1[r] + bv;
        yr[32] = acc2[r] + bv;
        yr[48] = acc3[r] + bv;
    }
}

// ---------------- BN stats: one block per channel, deterministic LDS reduction ----------------
__global__ void k_bn_stats(const float* __restrict__ Y, const float* __restrict__ gamma,
                           const float* __restrict__ beta, float* __restrict__ stats,
                           int O, int Qn) {
    const int o = blockIdx.x;
    const int t = threadIdx.x;
    float s = 0.f, s2 = 0.f;
    for (int b = 0; b < NB; ++b) {
        const float* row = Y + ((size_t)b * O + o) * Qn;
        for (int i = t; i < Qn; i += blockDim.x) {
            float v = row[i];
            s  += v;
            s2 += v * v;
        }
    }
    __shared__ float ls[256], ls2[256];
    ls[t] = s; ls2[t] = s2;
    __syncthreads();
    for (int st = 128; st > 0; st >>= 1) {
        if (t < st) { ls[t] += ls[t + st]; ls2[t] += ls2[t + st]; }
        __syncthreads();
    }
    if (t == 0) {
        float n = (float)(NB * Qn);
        float mu = ls[0] / n;
        float var = ls2[0] / n - mu * mu;
        float rstd = rsqrtf(var + 1e-5f);
        float sc = gamma[o] * rstd;
        stats[2 * o]     = sc;
        stats[2 * o + 1] = beta[o] - mu * sc;
    }
}

// ------- BN+ReLU applied, emitted as bf16 transposed (B, Q, O) for next GEMM -------
__global__ void k_bn_relu_to_bf16T(const float* __restrict__ Y, const float* __restrict__ stats,
                                   __bf16* __restrict__ Xt, int O, int Qn) {
    int q = blockIdx.x * blockDim.x + threadIdx.x;
    int o = blockIdx.y;
    int b = blockIdx.z;
    float sc = stats[2 * o], sh = stats[2 * o + 1];
    float v = Y[((size_t)b * O + o) * Qn + q];
    v = fmaxf(fmaf(v, sc, sh), 0.f);
    Xt[((size_t)b * Qn + q) * O + o] = (__bf16)v;
}

// ---------------- BN+ReLU in place on final f32 output ----------------
__global__ void k_bn_relu_inplace(float* __restrict__ Y, const float* __restrict__ stats,
                                  int O, int Qn) {
    int q = blockIdx.x * blockDim.x + threadIdx.x;
    int o = blockIdx.y;
    int b = blockIdx.z;
    float sc = stats[2 * o], sh = stats[2 * o + 1];
    size_t idx = ((size_t)b * O + o) * Qn + q;
    Y[idx] = fmaxf(fmaf(Y[idx], sc, sh), 0.f);
}

extern "C" void kernel_launch(void* const* d_in, const int* in_sizes, int n_in,
                              void* d_out, int out_size, void* d_ws, size_t ws_size,
                              hipStream_t stream) {
    const float* q_points = (const float*)d_in[0];
    const float* s_points = (const float*)d_in[1];
    const float* q_feats  = (const float*)d_in[2];
    const float* s_feats  = (const float*)d_in[3];
    const float* W1     = (const float*)d_in[4];
    const float* b1     = (const float*)d_in[5];
    const float* gamma1 = (const float*)d_in[6];
    const float* beta1  = (const float*)d_in[7];
    const float* W2     = (const float*)d_in[8];
    const float* b2     = (const float*)d_in[9];
    const float* gamma2 = (const float*)d_in[10];
    const float* beta2  = (const float*)d_in[11];
    float* out = (float*)d_out;

    // workspace carve-up (all offsets 256B aligned)
    char* ws = (char*)d_ws;
    size_t off = 0;
    auto carve = [&](size_t bytes) { void* p = ws + off; off += (bytes + 255) & ~(size_t)255; return p; };
    int*    nn_idx = (int*)   carve((size_t)NB * NQ * 3 * 4);
    float*  nn_w   = (float*) carve((size_t)NB * NQ * 3 * 4);
    __bf16* W1bf   = (__bf16*)carve((size_t)O1 * CIN * 2);
    __bf16* W2bf   = (__bf16*)carve((size_t)O2 * O1 * 2);
    __bf16* Xt     = (__bf16*)carve((size_t)NB * NQ * CIN * 2);
    __bf16* X2t    = (__bf16*)carve((size_t)NB * NQ * O1 * 2);
    float*  stats  = (float*) carve((size_t)256 * 2 * 4);
    (void)ws_size;

    // 1) weights -> bf16
    k_f32_to_bf16<<<(O1 * CIN + 255) / 256, 256, 0, stream>>>(W1, W1bf, O1 * CIN);
    k_f32_to_bf16<<<(O2 * O1 + 255) / 256, 256, 0, stream>>>(W2, W2bf, O2 * O1);

    // 2) three-NN + normalized inverse-distance weights
    k_three_nn<<<dim3(NQ / 256, NB), 256, 0, stream>>>(q_points, s_points, nn_idx, nn_w);

    // 3) interpolation + concat -> bf16 (B, Q, 384)
    k_interp_concat<<<dim3(NQ, NB), CIN, 0, stream>>>(q_feats, s_feats, nn_idx, nn_w, Xt);

    // 4) GEMM1 (WMMA bf16): pre-BN y1 -> d_out
    k_gemm_bf16<<<dim3(NQ / 64, O1 / 16, NB), 32, 0, stream>>>(W1bf, Xt, b1, out, O1, CIN, NQ);

    // 5) BN1 stats
    k_bn_stats<<<O1, 256, 0, stream>>>(out, gamma1, beta1, stats, O1, NQ);

    // 6) BN1 + ReLU -> bf16 transposed activations
    k_bn_relu_to_bf16T<<<dim3(NQ / 256, O1, NB), 256, 0, stream>>>(out, stats, X2t, O1, NQ);

    // 7) GEMM2 (WMMA bf16): pre-BN y2 -> d_out (overwrite)
    k_gemm_bf16<<<dim3(NQ / 64, O2 / 16, NB), 32, 0, stream>>>(W2bf, X2t, b2, out, O2, O1, NQ);

    // 8) BN2 stats
    k_bn_stats<<<O2, 256, 0, stream>>>(out, gamma2, beta2, stats, O2, NQ);

    // 9) BN2 + ReLU in place -> final output
    k_bn_relu_inplace<<<dim3(NQ / 256, O2, NB), 256, 0, stream>>>(out, stats, O2, NQ);

    (void)in_sizes; (void)n_in; (void)out_size;
}
